// LSTM_Encoder_66520453480925
// MI455X (gfx1250) — matrix operand order repaired and checked by reading, not compiled, not run on hardware
//
#include <hip/hip_runtime.h>
#include <hip/hip_bf16.h>

// ---------------------------------------------------------------------------
// BiLSTM encoder (2 layers, B=32, S=64, H=256/dir) + span features, CDNA5.
// All matmuls via v_wmma_f32_16x16x32_f16 (f16 inputs, f32 accumulate).
// ---------------------------------------------------------------------------

typedef _Float16 v16h __attribute__((ext_vector_type(16)));
typedef _Float16 v8h  __attribute__((ext_vector_type(8)));
typedef float    v8f  __attribute__((ext_vector_type(8)));

#define B_    32
#define S_    64
#define H_    256
#define G_    1024      // 4*H
#define KP0   448       // 428 padded to 32
#define KP1   512
#define LMAX_ 1953      // 63*62/2
#define TRAINV_ 40000

__device__ __forceinline__ v16h mk16(v8h lo, v8h hi) {
  return __builtin_shufflevector(lo, hi, 0,1,2,3,4,5,6,7,8,9,10,11,12,13,14,15);
}
__device__ __forceinline__ v8f wmma16(v16h a, v16h b, v8f c) {
  // D = A(16x32) * B(32x16) + C, f32 accumulate
  return __builtin_amdgcn_wmma_f32_16x16x32_f16(false, a, false, b, (short)0, c,
                                                false, false);
}
__device__ __forceinline__ float sigm(float x) { return 1.0f / (1.0f + __expf(-x)); }

// ---------------------------------------------------------------------------
// lens + output tail (lens-2 as float)
// ---------------------------------------------------------------------------
__global__ void lens_kernel(const int* __restrict__ word, int* __restrict__ slens,
                            float* __restrict__ out_tail) {
  int b = threadIdx.x;
  if (b >= B_) return;
  int c = 0;
  for (int s = 0; s < S_; ++s) c += (word[b * S_ + s] != 0);
  slens[b] = c;
  out_tail[b] = (float)(c - 2);
}

// ---------------------------------------------------------------------------
// zero the span output region
// ---------------------------------------------------------------------------
__global__ void zero_kernel(float4* __restrict__ p, long n4) {
  long i = (long)blockIdx.x * blockDim.x + threadIdx.x;
  long stride = (long)gridDim.x * blockDim.x;
  float4 z; z.x = 0.f; z.y = 0.f; z.z = 0.f; z.w = 0.f;
  for (; i < n4; i += stride) p[i] = z;
}

// ---------------------------------------------------------------------------
// weight f32 -> f16 with K padding.  grid = G rows, block = Kp
// ---------------------------------------------------------------------------
__global__ void convw_kernel(const float* __restrict__ W, _Float16* __restrict__ Wp,
                             int Kin, int Kp) {
  int g = blockIdx.x, k = threadIdx.x;
  Wp[(size_t)g * Kp + k] = (_Float16)(k < Kin ? W[(size_t)g * Kin + k] : 0.0f);
}

// ---------------------------------------------------------------------------
// embeddings -> f16 X0 (b,s,KP0), forward + padded-reverse
// ---------------------------------------------------------------------------
__device__ float embed_fetch(int b, int s, int k,
                             const int* word, const int* pos, const int* dep,
                             const float* ext, const float* wemb,
                             const float* pemb, const float* demb) {
  if (k >= 428) return 0.0f;
  int idx = b * S_ + s;
  if (k < 300) {
    int w = word[idx];
    int tr = (w >= TRAINV_) ? 1 : w;
    return ext[(size_t)w * 300 + k] + wemb[(size_t)tr * 300 + k];
  } else if (k < 364) {
    return pemb[pos[idx] * 64 + (k - 300)];
  } else {
    return demb[dep[idx] * 64 + (k - 364)];
  }
}

__global__ void embed_kernel(const int* __restrict__ word, const int* __restrict__ pos,
                             const int* __restrict__ dep, const float* __restrict__ ext,
                             const float* __restrict__ wemb, const float* __restrict__ pemb,
                             const float* __restrict__ demb, const int* __restrict__ lens,
                             _Float16* __restrict__ X0f, _Float16* __restrict__ X0r) {
  int t = blockIdx.x, b = t >> 6, s = t & 63, k = threadIdx.x;  // block = 448
  float v = embed_fetch(b, s, k, word, pos, dep, ext, wemb, pemb, demb);
  X0f[((size_t)(b * S_ + s)) * KP0 + k] = (_Float16)v;
  int L = lens[b];
  int ss = L - 1 - s; ss = ss < 0 ? 0 : (ss > S_ - 1 ? S_ - 1 : ss);
  float v2 = embed_fetch(b, ss, k, word, pos, dep, ext, wemb, pemb, demb);
  X0r[((size_t)(b * S_ + s)) * KP0 + k] = (_Float16)v2;
}

// ---------------------------------------------------------------------------
// input projection GEMM: XS[s,b,g] = X[b,s,:] @ W^T[:,g] + bias[g]
// grid = (S_, 16), block = 128 (4 waves).  Wave handles 1 n-tile x 2 m-tiles.
// ---------------------------------------------------------------------------
__global__ void gemm_xs_kernel(const _Float16* __restrict__ X,   // (B,S,Kp) f16
                               const _Float16* __restrict__ Wp,  // (G,Kp)  f16
                               const float* __restrict__ bias,   // (G)
                               float* __restrict__ XS,           // (S,B,G)
                               int Kp) {
  const int s = blockIdx.x;
  const int wave = threadIdx.x >> 5;
  const int lane = threadIdx.x & 31;
  const int nl = lane & 15, g2 = lane >> 4;
  const int ntile = blockIdx.y * 4 + wave;
  const int n = ntile * 16 + nl;

  v8f acc0, acc1;
  float bv = bias[n];
#pragma unroll
  for (int r = 0; r < 8; ++r) { acc0[r] = bv; acc1[r] = bv; }

  const _Float16* xrow0 = X + ((size_t)(nl * S_ + s)) * Kp;          // b = nl
  const _Float16* xrow1 = X + ((size_t)((16 + nl) * S_ + s)) * Kp;   // b = 16+nl
  const _Float16* wrow  = Wp + (size_t)n * Kp + 16 * g2;

  for (int k0 = 0; k0 < Kp; k0 += 32) {
    v8h a0lo = *(const v8h*)(xrow0 + k0 + 16 * g2);
    v8h a0hi = *(const v8h*)(xrow0 + k0 + 16 * g2 + 8);
    v8h a1lo = *(const v8h*)(xrow1 + k0 + 16 * g2);
    v8h a1hi = *(const v8h*)(xrow1 + k0 + 16 * g2 + 8);
    v16h a0 = mk16(a0lo, a0hi);
    v16h a1 = mk16(a1lo, a1hi);
    v16h bf = *(const v16h*)(wrow + k0);
    acc0 = wmma16(a0, bf, acc0);
    acc1 = wmma16(a1, bf, acc1);
  }
#pragma unroll
  for (int r = 0; r < 8; ++r) {
    int b0 = r + 8 * g2;
    int b1 = 16 + r + 8 * g2;
    XS[((size_t)(s * B_ + b0)) * G_ + n] = acc0[r];
    XS[((size_t)(s * B_ + b1)) * G_ + n] = acc1[r];
  }
}

// ---------------------------------------------------------------------------
// LSTM scan.  grid.x = 2 (fwd/bwd), block = 1024 (32 waves -> 8 waves/SIMD).
// Wave (wslot, m): m = wave/16 picks the 16-row batch tile, wslot = wave%16
// owns h columns [16*wslot, 16*wslot+16).  Each wave computes its i,f,g,o
// tiles (4 x 8 k-step WMMAs), does the nonlinear update locally in registers,
// writes h (f16) back to LDS.  xs is loaded up front and added after the
// GEMM so the scattered loads overlap the whole WMMA K-loop.
// ---------------------------------------------------------------------------
__global__ void __launch_bounds__(1024)
lstm_scan_kernel(const float* __restrict__ XSf, const float* __restrict__ XSb,
                 const _Float16* __restrict__ Whhf, const _Float16* __restrict__ Whhb,
                 const int* __restrict__ lens,
                 float* __restrict__ YSf, float* __restrict__ YSb) {
  const int dir = blockIdx.x;
  const float*    XS  = dir ? XSb  : XSf;
  const _Float16* Whh = dir ? Whhb : Whhf;
  float*          YS  = dir ? YSb  : YSf;

  __shared__ __align__(16) _Float16 hbuf[B_][264];  // 264 = 256 + pad (bank skew)
  __shared__ int slens[B_];

  const int wave  = threadIdx.x >> 5;
  const int lane  = threadIdx.x & 31;
  const int nl    = lane & 15, g2 = lane >> 4;
  const int wslot = wave & 15;
  const int m     = wave >> 4;              // batch tile (rows m*16 .. m*16+15)
  const int ncol  = wslot * 16 + nl;        // owned h column
  const int rowb  = m * 16;

  for (int i = threadIdx.x; i < B_ * 264; i += blockDim.x)
    (&hbuf[0][0])[i] = (_Float16)0.0f;
  if (threadIdx.x < B_) slens[threadIdx.x] = lens[threadIdx.x];
  __syncthreads();

  int lenreg[8];
#pragma unroll
  for (int r = 0; r < 8; ++r) lenreg[r] = slens[rowb + r + 8 * g2];

  v8f c = {};

  for (int s = 0; s < S_; ++s) {
    // ---- xs loads issued early; consumed after the K-loop ----
    float xg[4][8];
#pragma unroll
    for (int gt = 0; gt < 4; ++gt) {
      int n = gt * 256 + ncol;
#pragma unroll
      for (int r = 0; r < 8; ++r) {
        int b = rowb + r + 8 * g2;
        xg[gt][r] = XS[((size_t)(s * B_ + b)) * G_ + n];
      }
    }
    if (s + 1 < S_)  // warm L2/L0 for next step's xs slice
      __builtin_prefetch(XS + ((size_t)((s + 1) * B_)) * G_ + threadIdx.x, 0, 1);

    // ---- gates = h @ Whh^T  (K = 256, 8 k-steps of 32) ----
    v8f acc[4] = {{}, {}, {}, {}};
#pragma unroll
    for (int kk = 0; kk < 8; ++kk) {
      int k0 = kk * 32;
      v8h alo = *(const v8h*)&hbuf[rowb + nl][k0 + 16 * g2];
      v8h ahi = *(const v8h*)&hbuf[rowb + nl][k0 + 16 * g2 + 8];
      v16h a = mk16(alo, ahi);
#pragma unroll
      for (int gt = 0; gt < 4; ++gt) {
        int n = gt * 256 + ncol;
        v16h bf = *(const v16h*)(Whh + (size_t)n * H_ + k0 + 16 * g2);
        acc[gt] = wmma16(a, bf, acc[gt]);
      }
    }
    __syncthreads();  // all hbuf reads done before updates

    // ---- nonlinear update, masked carry, write h + ys ----
#pragma unroll
    for (int r = 0; r < 8; ++r) {
      int b = rowb + r + 8 * g2;
      bool live = s < lenreg[r];
      float i_ = acc[0][r] + xg[0][r];
      float f_ = acc[1][r] + xg[1][r];
      float g_ = acc[2][r] + xg[2][r];
      float o_ = acc[3][r] + xg[3][r];
      float cn = sigm(f_) * c[r] + sigm(i_) * tanhf(g_);
      float hn = sigm(o_) * tanhf(cn);
      if (live) {
        c[r] = cn;
        hbuf[b][ncol] = (_Float16)hn;
      }
      YS[((size_t)(s * B_ + b)) * H_ + ncol] = live ? hn : 0.0f;
    }
    __syncthreads();  // h updates visible before next step's reads
  }
}

// ---------------------------------------------------------------------------
// build layer-1 input: X1f[b,s,:] = [of(b,s), reversed-masked ob(b,s)] (f16)
// and its padded-reverse X1r.  grid = B*S, block = 512.
// ---------------------------------------------------------------------------
__global__ void pack1_kernel(const float* __restrict__ ys0f, const float* __restrict__ ys0b,
                             const int* __restrict__ lens,
                             _Float16* __restrict__ X1f, _Float16* __restrict__ X1r) {
  int t = blockIdx.x, b = t >> 6, s = t & 63, k = threadIdx.x;
  int L = lens[b];
  float v;
  if (k < 256) v = ys0f[((size_t)(s * B_ + b)) * H_ + k];
  else         v = (s < L) ? ys0b[((size_t)((L - 1 - s) * B_ + b)) * H_ + (k - 256)] : 0.0f;
  X1f[((size_t)(b * S_ + s)) * KP1 + k] = (_Float16)v;

  int ss = L - 1 - s; ss = ss < 0 ? 0 : (ss > S_ - 1 ? S_ - 1 : ss);
  float v2;
  if (k < 256) v2 = ys0f[((size_t)(ss * B_ + b)) * H_ + k];
  else         v2 = (ss < L) ? ys0b[((size_t)((L - 1 - ss) * B_ + b)) * H_ + (k - 256)] : 0.0f;
  X1r[((size_t)(b * S_ + s)) * KP1 + k] = (_Float16)v2;
}

// ---------------------------------------------------------------------------
// span features: out[b, p(i,j), :] = [fh[b,j]-fh[b,i], bh[b,i+1]-bh[b,j+1]]
// p(i,j) = i*(L-2) - i*(i-1)/2 + (j-i-1) for i < j <= L-2.
// grid = (63, 32), block = 512 (thread = feature).
// ---------------------------------------------------------------------------
__global__ void span_kernel(const float* __restrict__ ys1f, const float* __restrict__ ys1b,
                            const int* __restrict__ lens, float* __restrict__ out) {
  int i = blockIdx.x;
  int b = blockIdx.y;
  int k = threadIdx.x;
  int L = lens[b];
  if (i > L - 3) return;
  int base = i * (L - 2) - (i * (i - 1)) / 2;
  float ai;
  if (k < 256) ai = ys1f[((size_t)(i * B_ + b)) * H_ + k];
  else         ai = ys1b[((size_t)((L - 2 - i) * B_ + b)) * H_ + (k - 256)];  // bh[b,i+1]
  for (int j = i + 1; j <= L - 2; ++j) {
    int p = base + (j - i - 1);
    float v;
    if (k < 256) v = ys1f[((size_t)(j * B_ + b)) * H_ + k] - ai;
    else         v = ai - ys1b[((size_t)((L - 2 - j) * B_ + b)) * H_ + (k - 256)];
    out[(((size_t)b * LMAX_ + p)) * 512 + k] = v;
  }
}

// ---------------------------------------------------------------------------
// host launcher
// ---------------------------------------------------------------------------
extern "C" void kernel_launch(void* const* d_in, const int* in_sizes, int n_in,
                              void* d_out, int out_size, void* d_ws, size_t ws_size,
                              hipStream_t stream) {
  const int*   word  = (const int*)d_in[0];
  const int*   pos   = (const int*)d_in[1];
  const int*   dep   = (const int*)d_in[2];
  const float* ext   = (const float*)d_in[3];
  const float* wemb  = (const float*)d_in[4];
  const float* pemb  = (const float*)d_in[5];
  const float* demb  = (const float*)d_in[6];
  const float* Wih0f = (const float*)d_in[7];
  const float* Whh0f = (const float*)d_in[8];
  const float* b0f   = (const float*)d_in[9];
  const float* Wih0b = (const float*)d_in[10];
  const float* Whh0b = (const float*)d_in[11];
  const float* b0b   = (const float*)d_in[12];
  const float* Wih1f = (const float*)d_in[13];
  const float* Whh1f = (const float*)d_in[14];
  const float* b1f   = (const float*)d_in[15];
  const float* Wih1b = (const float*)d_in[16];
  const float* Whh1b = (const float*)d_in[17];
  const float* b1b   = (const float*)d_in[18];
  float* out = (float*)d_out;

  char* ws = (char*)d_ws;
  size_t off = 0;
  auto alloc = [&](size_t bytes) -> char* {
    char* p = ws + off;
    off = (off + bytes + 255) & ~(size_t)255;
    return p;
  };

  int*       slens  = (int*)alloc(B_ * sizeof(int));
  _Float16*  X0f    = (_Float16*)alloc((size_t)B_ * S_ * KP0 * 2);
  _Float16*  X0r    = (_Float16*)alloc((size_t)B_ * S_ * KP0 * 2);
  _Float16*  W0fP   = (_Float16*)alloc((size_t)G_ * KP0 * 2);
  _Float16*  W0bP   = (_Float16*)alloc((size_t)G_ * KP0 * 2);
  _Float16*  U0fP   = (_Float16*)alloc((size_t)G_ * H_ * 2);
  _Float16*  U0bP   = (_Float16*)alloc((size_t)G_ * H_ * 2);
  _Float16*  W1fP   = (_Float16*)alloc((size_t)G_ * KP1 * 2);
  _Float16*  W1bP   = (_Float16*)alloc((size_t)G_ * KP1 * 2);
  _Float16*  U1fP   = (_Float16*)alloc((size_t)G_ * H_ * 2);
  _Float16*  U1bP   = (_Float16*)alloc((size_t)G_ * H_ * 2);
  float*     XSf    = (float*)alloc((size_t)S_ * B_ * G_ * 4);
  float*     XSb    = (float*)alloc((size_t)S_ * B_ * G_ * 4);
  float*     ys0f   = (float*)alloc((size_t)S_ * B_ * H_ * 4);
  float*     ys0b   = (float*)alloc((size_t)S_ * B_ * H_ * 4);
  float*     ys1f   = (float*)alloc((size_t)S_ * B_ * H_ * 4);
  float*     ys1b   = (float*)alloc((size_t)S_ * B_ * H_ * 4);
  _Float16*  X1f    = (_Float16*)alloc((size_t)B_ * S_ * KP1 * 2);
  _Float16*  X1r    = (_Float16*)alloc((size_t)B_ * S_ * KP1 * 2);
  (void)ws_size; (void)in_sizes; (void)n_in; (void)out_size;

  float* out_tail = out + (size_t)B_ * LMAX_ * 512;

  // 1. lengths (+ lens-2 tail output)
  lens_kernel<<<1, 32, 0, stream>>>(word, slens, out_tail);

  // 2. zero span region (B*LMAX*512 floats = 7,999,488 float4)
  zero_kernel<<<4096, 256, 0, stream>>>((float4*)out, 7999488L);

  // 3. weight conversion f32 -> f16 (with K padding)
  convw_kernel<<<G_, KP0, 0, stream>>>(Wih0f, W0fP, 428, KP0);
  convw_kernel<<<G_, KP0, 0, stream>>>(Wih0b, W0bP, 428, KP0);
  convw_kernel<<<G_, H_,  0, stream>>>(Whh0f, U0fP, H_, H_);
  convw_kernel<<<G_, H_,  0, stream>>>(Whh0b, U0bP, H_, H_);
  convw_kernel<<<G_, KP1, 0, stream>>>(Wih1f, W1fP, 512, KP1);
  convw_kernel<<<G_, KP1, 0, stream>>>(Wih1b, W1bP, 512, KP1);
  convw_kernel<<<G_, H_,  0, stream>>>(Whh1f, U1fP, H_, H_);
  convw_kernel<<<G_, H_,  0, stream>>>(Whh1b, U1bP, H_, H_);

  // 4. embeddings (forward + padded-reverse), f16
  embed_kernel<<<B_ * S_, KP0, 0, stream>>>(word, pos, dep, ext, wemb, pemb, demb,
                                            slens, X0f, X0r);

  // 5. layer-0 input projections (WMMA GEMMs)
  gemm_xs_kernel<<<dim3(S_, 16), 128, 0, stream>>>(X0f, W0fP, b0f, XSf, KP0);
  gemm_xs_kernel<<<dim3(S_, 16), 128, 0, stream>>>(X0r, W0bP, b0b, XSb, KP0);

  // 6. layer-0 recurrent scan (both directions concurrently)
  lstm_scan_kernel<<<2, 1024, 0, stream>>>(XSf, XSb, U0fP, U0bP, slens, ys0f, ys0b);

  // 7. layer-1 input (concat + reversal), f16
  pack1_kernel<<<B_ * S_, KP1, 0, stream>>>(ys0f, ys0b, slens, X1f, X1r);

  // 8. layer-1 input projections
  gemm_xs_kernel<<<dim3(S_, 16), 128, 0, stream>>>(X1f, W1fP, b1f, XSf, KP1);
  gemm_xs_kernel<<<dim3(S_, 16), 128, 0, stream>>>(X1r, W1bP, b1b, XSb, KP1);

  // 9. layer-1 recurrent scan
  lstm_scan_kernel<<<2, 1024, 0, stream>>>(XSf, XSb, U1fP, U1bP, slens, ys1f, ys1b);

  // 10. span features -> x_span
  span_kernel<<<dim3(S_ - 1, B_), 512, 0, stream>>>(ys1f, ys1b, slens, out);
}
